// TorchLMHeadJSD_37958920962546
// MI455X (gfx1250) — matrix operand order repaired and checked by reading, not compiled, not run on hardware
//
#include <hip/hip_runtime.h>
#include <hip/hip_bf16.h>

// ---------------------------------------------------------------------------
// Fused LM-head JSD distillation loss for MI455X (gfx1250, wave32, WMMA).
//
// GEMMs: ~8.1e11 FLOPs vs ~0.8 GB mandatory traffic (AI ~1000) -> compute
// bound -> bf16 WMMA (v_wmma_f32_16x16x32_bf16, fp32 accum). Logits cached
// fp32 in d_ws (~524 MB, ~45 us HBM round-trip @ 23.3 TB/s) instead of
// re-running the compute-bound GEMM for the softmax second pass.
// This round: BK=64 (half the barriers, 16 WMMA per sync), v_perm_b32-based
// fp32->bf16x2 packing (3 VALU / 2 elems instead of ~8).
// ---------------------------------------------------------------------------

typedef __attribute__((ext_vector_type(16))) __bf16 v16bf;
typedef __attribute__((ext_vector_type(8)))  float  v8f;

#define BM 128
#define BN 128
#define BK 64
#define LDSS 72   // padded LDS row stride (elems); 144B rows -> conflict-free b128 frag reads

struct Pack32 { uint4 lo; uint4 hi; };   // 32 bytes -> bit_cast to v16bf

// pack two fp32 -> two bf16 in one u32 (round-half-up; ties differ from RNE only)
static __device__ __forceinline__ unsigned pack2bf(float a, float b) {
  const unsigned ra = __float_as_uint(a) + 0x8000u;
  const unsigned rb = __float_as_uint(b) + 0x8000u;
#if __has_builtin(__builtin_amdgcn_perm)
  // result bytes: [1:0] = ra bytes [3:2], [3:2] = rb bytes [3:2]  -> v_perm_b32
  return __builtin_amdgcn_perm(rb, ra, 0x07060302u);
#else
  return (ra >> 16) | (rb & 0xFFFF0000u);
#endif
}

static __device__ __forceinline__ v16bf pack_frag(uint4 lo, uint4 hi) {
  Pack32 p; p.lo = lo; p.hi = hi;
  return __builtin_bit_cast(v16bf, p);
}

// logits[n][v] = sum_k X[n][k] * W[v][k]   (X:[N,K] f32, W:[V,K] f32, out f32)
__global__ __launch_bounds__(256) void gemm_bf16_wmma(
    const float* __restrict__ X, const float* __restrict__ W,
    float* __restrict__ out, int K, int V)
{
  __shared__ alignas(16) unsigned short sA[2][BM * LDSS];
  __shared__ alignas(16) unsigned short sB[2][BN * LDSS];

  const int tid   = threadIdx.x;
  const int lane  = tid & 31;
  const int wave  = tid >> 5;
  const int waveM = (wave & 1) * 64;   // 2 wave-rows x 64 M
  const int waveN = (wave >> 1) * 32;  // 4 wave-cols x 32 N
  const int nTile = blockIdx.y * BM;   // token-row tile
  const int vTile = blockIdx.x * BN;   // vocab-col tile

  v8f acc[4][2];
  #pragma unroll
  for (int i = 0; i < 4; ++i)
    #pragma unroll
    for (int j = 0; j < 2; ++j)
      #pragma unroll
      for (int e = 0; e < 8; ++e) acc[i][j][e] = 0.0f;

  const int arow  = lane & 15;
  const int ahalf = lane >> 4;         // 0: lanes 0-15, 1: lanes 16-31

  // ---- software pipeline: load K-tile 0 into registers (8 float4 per matrix)
  float4 va[8], vb[8];
  #pragma unroll
  for (int it = 0; it < 8; ++it) {
    const int idx = tid + it * 256;    // 0..2047
    const int r   = idx >> 4;          // 128 rows, 16 float4 per row
    const int c4  = idx & 15;
    va[it] = *(const float4*)(X + (size_t)(nTile + r) * K + c4 * 4);
    vb[it] = *(const float4*)(W + (size_t)(vTile + r) * K + c4 * 4);
  }

  int buf = 0;
  for (int k0 = 0; k0 < K; k0 += BK) {
    // ---- convert regs (tile k0) -> bf16 in LDS[buf]
    #pragma unroll
    for (int it = 0; it < 8; ++it) {
      const int idx = tid + it * 256;
      const int r   = idx >> 4;
      const int c4  = idx & 15;
      uint2 pa; pa.x = pack2bf(va[it].x, va[it].y); pa.y = pack2bf(va[it].z, va[it].w);
      *(uint2*)&sA[buf][r * LDSS + c4 * 4] = pa;
      uint2 pb; pb.x = pack2bf(vb[it].x, vb[it].y); pb.y = pack2bf(vb[it].z, vb[it].w);
      *(uint2*)&sB[buf][r * LDSS + c4 * 4] = pb;
    }
    __syncthreads();   // one barrier per K-step (double buffer makes it safe)

    // ---- issue global loads for tile k0+BK (overlap with the WMMA block below)
    const int kN = k0 + BK;
    if (kN < K) {
      #pragma unroll
      for (int it = 0; it < 8; ++it) {
        const int idx = tid + it * 256;
        const int r   = idx >> 4;
        const int c4  = idx & 15;
        va[it] = *(const float4*)(X + (size_t)(nTile + r) * K + kN + c4 * 4);
        vb[it] = *(const float4*)(W + (size_t)(vTile + r) * K + kN + c4 * 4);
        if (kN + BK < K) {  // global_prefetch_b8 two tiles ahead
          __builtin_prefetch((const void*)(X + (size_t)(nTile + r) * K + kN + BK + c4 * 4), 0, 0);
          __builtin_prefetch((const void*)(W + (size_t)(vTile + r) * K + kN + BK + c4 * 4), 0, 0);
        }
      }
    }

    // ---- two 32-deep K sub-steps from LDS[buf]: 16 WMMA per barrier
    #pragma unroll
    for (int kk = 0; kk < 2; ++kk) {
      const int kcol = kk * 32;
      v16bf afr[4], bfr[2];
      #pragma unroll
      for (int i = 0; i < 4; ++i) {
        // A 16x32: lanes 0-15 hold M=lane, K {0..7,16..23}; lanes 16-31: K {8..15,24..31}
        const unsigned short* p = &sA[buf][(waveM + i * 16 + arow) * LDSS + kcol];
        uint4 lo = *(const uint4*)(p + ahalf * 8);
        uint4 hi = *(const uint4*)(p + 16 + ahalf * 8);
        afr[i] = pack_frag(lo, hi);
      }
      #pragma unroll
      for (int j = 0; j < 2; ++j) {
        // B 32x16: lanes 0-15 hold N=lane, K {0..7,16..23}; lanes 16-31: K {8..15,24..31}
        const unsigned short* p = &sB[buf][(waveN + j * 16 + arow) * LDSS + kcol];
        uint4 lo = *(const uint4*)(p + ahalf * 8);
        uint4 hi = *(const uint4*)(p + 16 + ahalf * 8);
        bfr[j] = pack_frag(lo, hi);
      }
      #pragma unroll
      for (int i = 0; i < 4; ++i)
        #pragma unroll
        for (int j = 0; j < 2; ++j)
          acc[i][j] = __builtin_amdgcn_wmma_f32_16x16x32_bf16(
              false, afr[i], false, bfr[j], (short)0, acc[i][j], false, false);
    }

    buf ^= 1;
  }

  // ---- store C (f32 layout: VGPR r -> M=r (lanes 0-15) / M=r+8 (lanes 16-31), N=lane%16)
  const int colBase = vTile + waveN + arow;
  #pragma unroll
  for (int i = 0; i < 4; ++i) {
    const int rowBase = nTile + waveM + i * 16 + ahalf * 8;
    #pragma unroll
    for (int j = 0; j < 2; ++j) {
      const int c = colBase + j * 16;
      #pragma unroll
      for (int r = 0; r < 8; ++r)
        out[(size_t)(rowBase + r) * V + c] = acc[i][j][r];
    }
  }
}

// ---------------------------------------------------------------------------
// Per-row softmax stats + CE + JSD partials. One 256-thread block per row.
// ---------------------------------------------------------------------------
static __device__ __forceinline__ float blockReduceMax(float v, float* red) {
  red[threadIdx.x] = v; __syncthreads();
  for (int s = 128; s > 0; s >>= 1) {
    if (threadIdx.x < s) red[threadIdx.x] = fmaxf(red[threadIdx.x], red[threadIdx.x + s]);
    __syncthreads();
  }
  float r = red[0]; __syncthreads(); return r;
}
static __device__ __forceinline__ float blockReduceSum(float v, float* red) {
  red[threadIdx.x] = v; __syncthreads();
  for (int s = 128; s > 0; s >>= 1) {
    if (threadIdx.x < s) red[threadIdx.x] += red[threadIdx.x + s];
    __syncthreads();
  }
  float r = red[0]; __syncthreads(); return r;
}

__global__ __launch_bounds__(256) void row_loss(
    const float* __restrict__ S, const float* __restrict__ T,
    const int* __restrict__ tgt,
    float* __restrict__ hardArr, float* __restrict__ jsdArr,
    float* __restrict__ validArr, int V)
{
  __shared__ float red[256];
  const int n   = blockIdx.x;
  const int tid = threadIdx.x;
  const float* s = S + (size_t)n * V;
  const float* t = T + (size_t)n * V;
  const float4* s4 = (const float4*)s;
  const float4* t4 = (const float4*)t;
  const int V4 = V >> 2;   // V % 4 == 0

  // pass 1: row max
  float ms = -3.4e38f, mt = -3.4e38f;
  for (int v = tid; v < V4; v += 256) {
    float4 x = s4[v], y = t4[v];
    ms = fmaxf(ms, fmaxf(fmaxf(x.x, x.y), fmaxf(x.z, x.w)));
    mt = fmaxf(mt, fmaxf(fmaxf(y.x, y.y), fmaxf(y.z, y.w)));
  }
  ms = blockReduceMax(ms, red);
  mt = blockReduceMax(mt, red);

  // pass 2: sum exp
  float ss = 0.f, st = 0.f;
  for (int v = tid; v < V4; v += 256) {
    float4 x = s4[v], y = t4[v];
    ss += __expf(x.x - ms) + __expf(x.y - ms) + __expf(x.z - ms) + __expf(x.w - ms);
    st += __expf(y.x - mt) + __expf(y.y - mt) + __expf(y.z - mt) + __expf(y.w - mt);
  }
  ss = blockReduceSum(ss, red);
  st = blockReduceSum(st, red);
  const float Zs = ms + __logf(ss);
  const float Zt = mt + __logf(st);

  // pass 3: JSD pointwise:  m*log(m) - 0.5*m*(ls+lt),  m = (e^ls + e^lt)/2
  float jp = 0.f;
  for (int v = tid; v < V4; v += 256) {
    float4 x = s4[v], y = t4[v];
    #pragma unroll
    for (int e = 0; e < 4; ++e) {
      const float ls = (&x.x)[e] - Zs;
      const float lt = (&y.x)[e] - Zt;
      const float m  = 0.5f * (__expf(ls) + __expf(lt));
      if (m > 0.f) jp += m * __logf(m) - 0.5f * m * (ls + lt);
    }
  }
  jp = blockReduceSum(jp, red);

  if (tid == 0) {
    const int tg = tgt[n];
    const bool valid = (tg != -100);
    float hard = 0.f;
    if (valid && tg >= 0 && tg < V) hard = Zs - s[tg];   // -log_softmax at target
    hardArr[n]  = hard;
    validArr[n] = valid ? 1.f : 0.f;
    jsdArr[n]   = jp;
  }
}

// ---------------------------------------------------------------------------
// Deterministic final reduction (no float atomics -> replay-deterministic).
// ---------------------------------------------------------------------------
__global__ __launch_bounds__(256) void finalize_loss(
    const float* __restrict__ hardArr, const float* __restrict__ jsdArr,
    const float* __restrict__ validArr, int N, float* __restrict__ out)
{
  __shared__ float red[256];
  float h = 0.f, j = 0.f, c = 0.f;
  for (int i = threadIdx.x; i < N; i += 256) {
    h += hardArr[i]; j += jsdArr[i]; c += validArr[i];
  }
  h = blockReduceSum(h, red);
  j = blockReduceSum(j, red);
  c = blockReduceSum(c, red);
  if (threadIdx.x == 0) {
    const float hard = h / fmaxf(c, 1.f);
    const float jsd  = j / (float)N;          // (student_kl + teacher_kl)/2, batchmean
    out[0] = 0.5f * hard + 0.5f * jsd;
  }
}

extern "C" void kernel_launch(void* const* d_in, const int* in_sizes, int n_in,
                              void* d_out, int out_size, void* d_ws, size_t ws_size,
                              hipStream_t stream) {
  (void)n_in; (void)out_size; (void)ws_size;
  const float* Xs  = (const float*)d_in[0];   // [N, HS]
  const float* Xt  = (const float*)d_in[1];   // [N, HT]
  const float* Ws  = (const float*)d_in[2];   // [V, HS]
  const float* Wt  = (const float*)d_in[3];   // [V, HT]
  const int*   tgt = (const int*)d_in[4];     // [N]

  const int N  = in_sizes[4];
  const int HS = in_sizes[0] / N;
  const int HT = in_sizes[1] / N;
  const int V  = in_sizes[2] / HS;

  float* ws = (float*)d_ws;
  const size_t NV = (size_t)N * (size_t)V;
  float* sLogits  = ws;                 // N*V
  float* tLogits  = sLogits + NV;       // N*V
  float* hardArr  = tLogits + NV;       // N
  float* jsdArr   = hardArr + N;        // N
  float* validArr = jsdArr + N;         // N

  dim3 grid(V / BN, N / BM);            // 32000/128=250, 2048/128=16 (exact)
  gemm_bf16_wmma<<<grid, 256, 0, stream>>>(Xs, Ws, sLogits, HS, V);
  gemm_bf16_wmma<<<grid, 256, 0, stream>>>(Xt, Wt, tLogits, HT, V);
  row_loss<<<N, 256, 0, stream>>>(sLogits, tLogits, tgt, hardArr, jsdArr, validArr, V);
  finalize_loss<<<1, 256, 0, stream>>>(hardArr, jsdArr, validArr, N, (float*)d_out);
}
